// VectorQuantizer_29738353557975
// MI455X (gfx1250) — compile-verified, matrix operand order checked by reading, not run on hardware
//
#include <hip/hip_runtime.h>
#include <hip/hip_bf16.h>

typedef _Float16     v16h  __attribute__((ext_vector_type(16)));
typedef float        v8f   __attribute__((ext_vector_type(8)));
typedef float        v4f   __attribute__((ext_vector_type(4)));
typedef unsigned int v4u   __attribute__((ext_vector_type(4)));
typedef unsigned int u32x4 __attribute__((ext_vector_type(4)));
typedef int          i32x8 __attribute__((ext_vector_type(8)));
typedef int          i32x4 __attribute__((ext_vector_type(4)));

#define DIM 64
#define WAVES_PER_BLOCK 8
#define ROWS_PER_WAVE 16
#define ROWS_PER_BLOCK (WAVES_PER_BLOCK * ROWS_PER_WAVE)

// Codebook staged in LDS: 1024 rows, each 32 dwords data + 4 dwords TDM pad
// => 36-dword (144 B) row stride: 16B-aligned for ds_load_b128, and
// 36 mod 64 banks walks all 16 column-lanes onto distinct banks.
#define CB_ROWS        1024
#define CB_ROW_DWORDS  32
#define CB_PAD_DWORDS  4
#define CB_STRIDE_H    ((CB_ROW_DWORDS + CB_PAD_DWORDS) * 2)   // 72 halves
#define CB_LDS_BYTES   (CB_ROWS * (CB_ROW_DWORDS + CB_PAD_DWORDS) * 4)  // 147456

union Frag {
    v16h h;
    v4u  q[2];
};

// ---------------------------------------------------------------------------
// Kernel 1: normalize codebook rows, convert to f16 into workspace (contiguous).
// ---------------------------------------------------------------------------
__global__ __launch_bounds__(256)
void vq_prep_codebook(const float* __restrict__ cb, _Float16* __restrict__ cn16, int K) {
    int row = blockIdx.x * blockDim.x + threadIdx.x;
    if (row >= K) return;
    const float* src = cb + (long)row * DIM;
    float v[DIM];
    float ss = 0.f;
    #pragma unroll
    for (int j = 0; j < DIM / 4; ++j) {
        v4f f = reinterpret_cast<const v4f*>(src)[j];
        v[4*j+0] = f[0]; v[4*j+1] = f[1]; v[4*j+2] = f[2]; v[4*j+3] = f[3];
        ss += f[0]*f[0] + f[1]*f[1] + f[2]*f[2] + f[3]*f[3];
    }
    float scale = 1.0f / fmaxf(sqrtf(ss), 1e-12f);
    _Float16* dst = cn16 + (long)row * DIM;
    #pragma unroll
    for (int j = 0; j < DIM; ++j) dst[j] = (_Float16)(v[j] * scale);
}

// ---------------------------------------------------------------------------
// Kernel 2: TDM-stage codebook to LDS once per block; per-wave 16-row tile:
// normalize x, WMMA score vs all K codes from LDS, argmax, gather, STE, loss.
// ---------------------------------------------------------------------------
__global__ __launch_bounds__(256)
void vq_main(const float* __restrict__ x, const float* __restrict__ cb,
             const _Float16* __restrict__ cn16,
             float* __restrict__ outq, float* __restrict__ outidx,
             float* __restrict__ partials, int ktiles)
{
    __shared__ v4u   smemRaw[CB_LDS_BYTES / 16];   // codebook region; x-staging reuses it
    __shared__ int   sIdx[WAVES_PER_BLOCK][ROWS_PER_WAVE];
    __shared__ float sPart[WAVES_PER_BLOCK];

    const int lane = threadIdx.x & 31;
    const int wv   = threadIdx.x >> 5;
    const int r    = lane & 15;   // row within wave tile (A/F) / code column (B/C)
    const int h    = lane >> 4;   // half selector
    const long rowBase = ((long)blockIdx.x * WAVES_PER_BLOCK + wv) * ROWS_PER_WAVE;

    // ---- Phase A: load 16 x-rows, compute norms, write normalized f16 to LDS
    const float* xr = x + (rowBase + r) * DIM + h * 32;
    float xv[32];
    float ss = 0.f;
    #pragma unroll
    for (int j = 0; j < 8; ++j) {
        v4f f = reinterpret_cast<const v4f*>(xr)[j];
        xv[4*j+0] = f[0]; xv[4*j+1] = f[1]; xv[4*j+2] = f[2]; xv[4*j+3] = f[3];
        ss += f[0]*f[0] + f[1]*f[1] + f[2]*f[2] + f[3]*f[3];
    }
    ss += __shfl_xor(ss, 16);                 // combine both row halves
    float scale = 1.0f / fmaxf(sqrtf(ss), 1e-12f);
    {
        _Float16* dst = ((_Float16*)smemRaw) + (wv * ROWS_PER_WAVE + r) * DIM + h * 32;
        #pragma unroll
        for (int j = 0; j < 32; ++j) dst[j] = (_Float16)(xv[j] * scale);
    }
    __syncthreads();

    // ---- Phase B: build A fragments (CDNA5 16-bit 16x32 A layout)
    // lane L<16: row L, K = {0..7, 16..23}; lane L+16: row L, K = {8..15, 24..31}
    const _Float16* arow = ((const _Float16*)smemRaw) + (wv * ROWS_PER_WAVE + r) * DIM;
    Frag a0, a1;
    a0.q[0] = *reinterpret_cast<const v4u*>(arow +      8 * h);
    a0.q[1] = *reinterpret_cast<const v4u*>(arow + 16 + 8 * h);
    a1.q[0] = *reinterpret_cast<const v4u*>(arow + 32 +      8 * h);
    a1.q[1] = *reinterpret_cast<const v4u*>(arow + 32 + 16 + 8 * h);
    __syncthreads();   // everyone done reading x-staging before TDM overwrites it

    // ---- Phase T: one TDM DMA stages the whole f16 codebook into LDS,
    // padding 4 dwords after each 32-dword row (row stride 144 B).
    if (wv == 0) {
        unsigned long long ga = (unsigned long long)(uintptr_t)cn16;
        unsigned ldsOff = (unsigned)(unsigned long long)(uintptr_t)&smemRaw[0]; // low 32 = LDS offset

        u32x4 g0;
        g0[0] = 1u;                                           // count=1, user descriptor
        g0[1] = ldsOff;                                       // lds_addr
        g0[2] = (unsigned)(ga & 0xFFFFFFFFu);                 // global_addr[31:0]
        g0[3] = (unsigned)((ga >> 32) & 0x01FFFFFFu) | (2u << 30); // addr[56:32] | type=2

        i32x8 g1 = {};
        // data_size=4B, pad_enable, pad_interval=32 dwords (enc 4), pad_amount=4 dwords (enc 3)
        g1[0] = (2 << 16) | (1 << 20) | (4 << 22) | (3 << 25);
        g1[1] = (int)((unsigned)(CB_ROW_DWORDS & 0xFFFF) << 16);        // tensor_dim0 lo
        g1[2] = (int)((unsigned)(CB_ROWS & 0xFFFF) << 16);              // dim0 hi=0 | tensor_dim1 lo
        g1[3] = (int)((unsigned)CB_ROW_DWORDS << 16);                   // dim1 hi=0 | tile_dim0
        g1[4] = (int)(CB_ROWS);                                         // tile_dim1 | tile_dim2=0
        g1[5] = CB_ROW_DWORDS;                                          // tensor_dim0_stride lo32
        g1[6] = 0;
        g1[7] = 0;
        i32x4 z4 = {};
        i32x8 z8 = {};
        __builtin_amdgcn_tensor_load_to_lds(g0, g1, z4, z4, z8, 0);
        __builtin_amdgcn_s_wait_tensorcnt(0);
    }
    __syncthreads();

    // ---- Phase C: sweep codebook tiles from LDS, running argmax of dots
    float best[8];
    int   bidx[8];
    #pragma unroll
    for (int i = 0; i < 8; ++i) { best[i] = -3.4e38f; bidx[i] = 0; }

    const _Float16* bp = ((const _Float16*)smemRaw) + (size_t)r * CB_STRIDE_H + 8 * h;
    #pragma unroll 4
    for (int t = 0; t < ktiles; ++t) {
        Frag b0, b1;
        b0.q[0] = *reinterpret_cast<const v4u*>(bp);
        b0.q[1] = *reinterpret_cast<const v4u*>(bp + 16);
        b1.q[0] = *reinterpret_cast<const v4u*>(bp + 32);
        b1.q[1] = *reinterpret_cast<const v4u*>(bp + 48);
        bp += 16 * CB_STRIDE_H;

        v8f acc = {};
        acc = __builtin_amdgcn_wmma_f32_16x16x32_f16(false, a0.h, false, b0.h,
                                                     (short)0, acc, false, false);
        acc = __builtin_amdgcn_wmma_f32_16x16x32_f16(false, a1.h, false, b1.h,
                                                     (short)0, acc, false, false);
        const int cand = t * 16 + r;   // this lane's code index (column N = lane%16)
        #pragma unroll
        for (int i = 0; i < 8; ++i) {
            float v = acc[i];
            if (v > best[i]) { best[i] = v; bidx[i] = cand; }  // strict > keeps lowest idx
        }
    }

    // ---- Phase D: argmax across the 16 column-lanes (per 16-lane group)
    #pragma unroll
    for (int off = 1; off < 16; off <<= 1) {
        #pragma unroll
        for (int i = 0; i < 8; ++i) {
            float ov = __shfl_xor(best[i], off);
            int   oi = __shfl_xor(bidx[i], off);
            if (ov > best[i] || (ov == best[i] && oi < bidx[i])) {
                best[i] = ov; bidx[i] = oi;
            }
        }
    }

    // C-layout: lanes 0-15 VGPR i -> row i; lanes 16-31 VGPR i -> row 8+i
    if (r == 0) {
        #pragma unroll
        for (int i = 0; i < 8; ++i) sIdx[wv][h * 8 + i] = bidx[i];
    }
    __syncthreads();

    // ---- Phase F: gather unnormalized code rows, STE output, loss partial
    const int code = sIdx[wv][r];
    const float* qrow = cb + (long)code * DIM + h * 32;
    float* orow = outq + (rowBase + r) * DIM + h * 32;
    float lsum = 0.f;
    #pragma unroll
    for (int j = 0; j < 8; ++j) {
        v4f q = reinterpret_cast<const v4f*>(qrow)[j];
        v4f o;
        #pragma unroll
        for (int c = 0; c < 4; ++c) {
            float xf = xv[4*j + c];
            float d  = q[c] - xf;        // stop_gradient(q - x) value
            lsum += d * d;
            o[c] = xf + d;               // x + (q - x), bit-faithful to reference
        }
        reinterpret_cast<v4f*>(orow)[j] = o;
    }
    if (lane < 16) outidx[rowBase + lane] = (float)sIdx[wv][lane];

    #pragma unroll
    for (int off = 1; off < 32; off <<= 1) lsum += __shfl_xor(lsum, off);
    if (lane == 0) sPart[wv] = lsum;
    __syncthreads();
    if (threadIdx.x == 0) {
        float tot = 0.f;
        #pragma unroll
        for (int w = 0; w < WAVES_PER_BLOCK; ++w) tot += sPart[w];  // fixed order
        partials[blockIdx.x] = tot;
    }
}

// ---------------------------------------------------------------------------
// Kernel 3: deterministic reduction of per-block partials -> both loss scalars
// ---------------------------------------------------------------------------
__global__ __launch_bounds__(256)
void vq_loss_reduce(const float* __restrict__ partials, int nb,
                    float* __restrict__ outloss, float invCount)
{
    __shared__ float s[256];
    float t = 0.f;
    for (int i = threadIdx.x; i < nb; i += 256) t += partials[i]; // fixed per-thread order
    s[threadIdx.x] = t;
    __syncthreads();
    for (int off = 128; off > 0; off >>= 1) {
        if ((int)threadIdx.x < off) s[threadIdx.x] += s[threadIdx.x + off];
        __syncthreads();
    }
    if (threadIdx.x == 0) {
        float m = s[0] * invCount;
        outloss[0] = m;   // codebook_loss
        outloss[1] = m;   // commitment_loss (identical value in reference)
    }
}

// ---------------------------------------------------------------------------
extern "C" void kernel_launch(void* const* d_in, const int* in_sizes, int n_in,
                              void* d_out, int out_size, void* d_ws, size_t ws_size,
                              hipStream_t stream) {
    const float* x  = (const float*)d_in[0];
    const float* cb = (const float*)d_in[1];
    const int N = in_sizes[0] / DIM;   // 262144
    const int K = in_sizes[1] / DIM;   // 1024

    _Float16* cn16   = (_Float16*)d_ws;
    float* partials  = (float*)((char*)d_ws + (size_t)K * DIM * sizeof(_Float16));

    float* out     = (float*)d_out;
    const long QN  = (long)N * DIM;
    float* outq    = out;
    float* outloss = out + QN;
    float* outidx  = out + QN + 2;

    const int nb = N / ROWS_PER_BLOCK;   // 2048 blocks

    vq_prep_codebook<<<(K + 255) / 256, 256, 0, stream>>>(cb, cn16, K);
    vq_main<<<nb, 256, 0, stream>>>(x, cb, cn16, outq, outidx, partials, K / 16);
    vq_loss_reduce<<<1, 256, 0, stream>>>(partials, nb, outloss,
                                          1.0f / (float)((long)N * DIM));
}
